// MultiscaleGlobalWaveGNN_V5_59064390255210
// MI455X (gfx1250) — matrix operand order repaired and checked by reading, not compiled, not run on hardware
//
#include <hip/hip_runtime.h>

#define BB 2
#define NN 40962
#define DD 128
#define HH 4
#define PP 100
#define SS 512
#define HD 32
#define MT (BB * PP * SS) /* 102400 */

typedef __bf16 v16bf __attribute__((ext_vector_type(16)));
typedef float  v8f   __attribute__((ext_vector_type(8)));

union BFrag { v16bf v; uint4 q[2]; };

// round-to-nearest-even fp32 -> bf16 bit pattern
__device__ __forceinline__ unsigned short f2bf(float f) {
    union { float f; unsigned u; } u; u.f = f;
    unsigned r = u.u + 0x7FFFu + ((u.u >> 16) & 1u);
    return (unsigned short)(r >> 16);
}

// A-fragment (16x32 bf16, M = lane%16): 16 consecutive K values per lane
__device__ __forceinline__ v16bf ldA(const unsigned short* p) {
    BFrag f; f.q[0] = *(const uint4*)p; f.q[1] = *(const uint4*)(p + 8); return f.v;
}
// B-fragment (32x16 bf16, N = lane%16): 16 consecutive K values per lane
__device__ __forceinline__ v16bf ldB(const unsigned short* p) {
    BFrag f; f.q[0] = *(const uint4*)p; f.q[1] = *(const uint4*)(p + 8); return f.v;
}
__device__ __forceinline__ v8f wmma_bf(v16bf a, v16bf b, v8f c) {
    return __builtin_amdgcn_wmma_f32_16x16x32_bf16(false, a, false, b, (short)0, c, false, false);
}

// ---------------- 1. weights fp32 -> bf16 (4 x 128x128, row-major [n][k]) ------------
__global__ __launch_bounds__(256) void k_wconv(const float* wq, const float* wk,
                                               const float* wv, const float* wo,
                                               unsigned short* dst) {
    int i = blockIdx.x * 256 + threadIdx.x;
    if (i >= 4 * DD * DD) return;
    int w = i >> 14, e = i & 16383;
    const float* src = (w == 0) ? wq : (w == 1) ? wk : (w == 2) ? wv : wo;
    dst[i] = f2bf(src[e]);
}

// ---------------- 2. gather px = x[:, patch_indices] -> bf16 ------------------------
__global__ __launch_bounds__(256) void k_gather(const float* x, const int* pidx,
                                                unsigned short* px) {
    int i = blockIdx.x * 256 + threadIdx.x;   // float4 granularity
    if (i >= MT * (DD / 4)) return;
    int m = i >> 5, c = (i & 31) << 2;
    int b = m / (PP * SS), ps = m % (PP * SS);
    int node = pidx[ps];
    float4 v = *(const float4*)(x + ((size_t)b * NN + node) * DD + c);
    ushort4 o; o.x = f2bf(v.x); o.y = f2bf(v.y); o.z = f2bf(v.z); o.w = f2bf(v.w);
    *(ushort4*)(px + (size_t)m * DD + c) = o;
}

// ---------------- 3. QKV GEMM: out = px @ W^T + b (bf16 in/out, fp32 acc) -----------
__global__ __launch_bounds__(256) void k_qkv(const unsigned short* __restrict__ px,
                                             const unsigned short* __restrict__ wb,
                                             const float* bq, const float* bk, const float* bv,
                                             unsigned short* qo, unsigned short* ko,
                                             unsigned short* vo) {
    const int wave = threadIdx.x >> 5, lane = threadIdx.x & 31;
    const int mlo = lane & 15, hi = lane >> 4;
    const int row0 = (blockIdx.x * 8 + wave) << 4;

    v16bf a[4];
#pragma unroll
    for (int kt = 0; kt < 4; ++kt)
        a[kt] = ldA(px + (size_t)(row0 + mlo) * DD + kt * 32 + (hi << 4));

#pragma unroll 1
    for (int widx = 0; widx < 3; ++widx) {
        const unsigned short* W = wb + widx * DD * DD;
        const float* bias = (widx == 0) ? bq : (widx == 1) ? bk : bv;
        unsigned short* out = (widx == 0) ? qo : (widx == 1) ? ko : vo;
        v8f c[8];
#pragma unroll
        for (int nt = 0; nt < 8; ++nt) c[nt] = v8f{0.f,0.f,0.f,0.f,0.f,0.f,0.f,0.f};
#pragma unroll
        for (int kt = 0; kt < 4; ++kt)
#pragma unroll
            for (int nt = 0; nt < 8; ++nt) {
                v16bf bfr = ldB(W + (size_t)(nt * 16 + mlo) * DD + kt * 32 + (hi << 4));
                c[nt] = wmma_bf(a[kt], bfr, c[nt]);
            }
#pragma unroll
        for (int nt = 0; nt < 8; ++nt) {
            int n = nt * 16 + mlo;
            float bsv = bias[n];
#pragma unroll
            for (int r = 0; r < 8; ++r) {
                int row = row0 + r + (hi << 3);
                out[(size_t)row * DD + n] = f2bf(c[nt][r] + bsv);
            }
        }
    }
}

// ---------------- 4. flash attention per (b,p,h): S=512, HD=32 ----------------------
__global__ __launch_bounds__(256) void k_attn(const unsigned short* __restrict__ qb,
                                              const unsigned short* __restrict__ kb,
                                              const unsigned short* __restrict__ vb,
                                              unsigned short* __restrict__ attb) {
    __shared__ unsigned short vT[HD * 520];     // v transposed [d][t], padded stride
    __shared__ unsigned short Pw[8][16 * 40];   // per-wave P bounce, padded stride

    const int id = blockIdx.x;
    const int h = id & 3;
    const int p = (id >> 2) % PP;
    const int b = id / (PP * HH);
    const size_t base = ((size_t)(b * PP + p) * SS) * DD + h * HD;

    // cooperative transpose of V into LDS
    for (int i = threadIdx.x; i < SS * HD; i += 256) {
        int t = i >> 5, d = i & 31;
        vT[d * 520 + t] = vb[base + (size_t)t * DD + d];
    }
    __syncthreads();

    const int wave = threadIdx.x >> 5, lane = threadIdx.x & 31;
    const int mlo = lane & 15, hi = lane >> 4;
    const float scale = 0.17677669529663687f;   // 1/sqrt(32)
    unsigned short* pw = &Pw[wave][0];

    for (int blk = wave; blk < SS / 16; blk += 8) {
        const int s0 = blk << 4;
        v16bf aq = ldA(qb + base + (size_t)(s0 + mlo) * DD + (hi << 4));

        float mr[8], lr[8];
        v8f o0 = v8f{0.f,0.f,0.f,0.f,0.f,0.f,0.f,0.f};
        v8f o1 = v8f{0.f,0.f,0.f,0.f,0.f,0.f,0.f,0.f};
#pragma unroll
        for (int r = 0; r < 8; ++r) { mr[r] = -3.0e38f; lr[r] = 0.f; }

        for (int t0 = 0; t0 < SS; t0 += 32) {
            v16bf bk0 = ldB(kb + base + (size_t)(t0 + mlo) * DD + (hi << 4));
            v16bf bk1 = ldB(kb + base + (size_t)(t0 + 16 + mlo) * DD + (hi << 4));
            v8f zc = v8f{0.f,0.f,0.f,0.f,0.f,0.f,0.f,0.f};
            v8f c0 = wmma_bf(aq, bk0, zc);
            v8f c1 = wmma_bf(aq, bk1, zc);

            float p0[8], p1[8];
#pragma unroll
            for (int r = 0; r < 8; ++r) {
                float s0v = c0[r] * scale, s1v = c1[r] * scale;
                float mx = fmaxf(s0v, s1v);
                mx = fmaxf(mx, __shfl_xor(mx, 1));
                mx = fmaxf(mx, __shfl_xor(mx, 2));
                mx = fmaxf(mx, __shfl_xor(mx, 4));
                mx = fmaxf(mx, __shfl_xor(mx, 8));
                float mn = fmaxf(mr[r], mx);
                float al = __expf(mr[r] - mn);
                float e0 = __expf(s0v - mn), e1 = __expf(s1v - mn);
                float rs = e0 + e1;
                rs += __shfl_xor(rs, 1);
                rs += __shfl_xor(rs, 2);
                rs += __shfl_xor(rs, 4);
                rs += __shfl_xor(rs, 8);
                lr[r] = lr[r] * al + rs;
                mr[r] = mn;
                o0[r] *= al; o1[r] *= al;
                p0[r] = e0; p1[r] = e1;
            }
            // bounce P (C-layout) through LDS to re-read as an A-fragment
#pragma unroll
            for (int r = 0; r < 8; ++r) {
                int rr = r + (hi << 3);
                pw[rr * 40 + mlo]      = f2bf(p0[r]);
                pw[rr * 40 + 16 + mlo] = f2bf(p1[r]);
            }
            asm volatile("s_wait_dscnt 0x0" ::: "memory");  // intra-wave LDS RAW
            v16bf ap  = ldA(pw + mlo * 40 + (hi << 4));
            v16bf bv0 = ldB(vT + (size_t)mlo * 520 + t0 + (hi << 4));
            v16bf bv1 = ldB(vT + (size_t)(16 + mlo) * 520 + t0 + (hi << 4));
            o0 = wmma_bf(ap, bv0, o0);
            o1 = wmma_bf(ap, bv1, o1);
        }
#pragma unroll
        for (int r = 0; r < 8; ++r) {
            float inv = 1.0f / lr[r];
            int row = s0 + r + (hi << 3);
            attb[base + (size_t)row * DD + mlo]      = f2bf(o0[r] * inv);
            attb[base + (size_t)row * DD + 16 + mlo] = f2bf(o1[r] * inv);
        }
    }
}

// ---------------- 5. Wo GEMM + mask + scatter-add into agg --------------------------
__global__ __launch_bounds__(256) void k_wo(const unsigned short* __restrict__ att,
                                            const unsigned short* __restrict__ wo_b,
                                            const float* bo, const int* pidx,
                                            const unsigned char* cmask,
                                            float* __restrict__ agg) {
    const int wave = threadIdx.x >> 5, lane = threadIdx.x & 31;
    const int mlo = lane & 15, hi = lane >> 4;
    const int row0 = (blockIdx.x * 8 + wave) << 4;

    v16bf a[4];
#pragma unroll
    for (int kt = 0; kt < 4; ++kt)
        a[kt] = ldA(att + (size_t)(row0 + mlo) * DD + kt * 32 + (hi << 4));

    v8f c[8];
#pragma unroll
    for (int nt = 0; nt < 8; ++nt) c[nt] = v8f{0.f,0.f,0.f,0.f,0.f,0.f,0.f,0.f};
#pragma unroll
    for (int kt = 0; kt < 4; ++kt)
#pragma unroll
        for (int nt = 0; nt < 8; ++nt) {
            v16bf bfr = ldB(wo_b + (size_t)(nt * 16 + mlo) * DD + kt * 32 + (hi << 4));
            c[nt] = wmma_bf(a[kt], bfr, c[nt]);
        }

#pragma unroll
    for (int r = 0; r < 8; ++r) {
        int row = row0 + r + (hi << 3);
        int ps = row % (PP * SS);
        int bb = row / (PP * SS);
        if (!cmask[ps]) continue;                       // masked row contributes zero
        int node = pidx[ps];
        float* dst = agg + ((size_t)bb * NN + node) * DD;
#pragma unroll
        for (int nt = 0; nt < 8; ++nt) {
            int n = nt * 16 + mlo;
            atomicAdd(dst + n, c[nt][r] + bo[n]);
        }
    }
}

// ---------------- 6. segment counts ------------------------------------------------
__global__ __launch_bounds__(256) void k_cnt(const int* pidx, const unsigned char* cm,
                                             float* cnt) {
    int i = blockIdx.x * 256 + threadIdx.x;
    if (i < PP * SS && cm[i]) atomicAdd(&cnt[pidx[i]], 1.0f);
}

// ---------------- 7. residual + LayerNorm ------------------------------------------
__global__ __launch_bounds__(128) void k_ln(const float* __restrict__ x,
                                            const float* __restrict__ agg,
                                            const float* __restrict__ cnt,
                                            const float* g, const float* bt,
                                            float* __restrict__ out) {
    __shared__ float red[4][2];
    const int row = blockIdx.x;
    const int d = threadIdx.x;
    const int n = row % NN;
    float c = cnt[n]; c = (c < 1.f) ? 1.f : c;
    float v = x[(size_t)row * DD + d] + agg[(size_t)row * DD + d] / c;

    float s = v, s2 = v * v;
#pragma unroll
    for (int m = 1; m < 32; m <<= 1) { s += __shfl_xor(s, m); s2 += __shfl_xor(s2, m); }
    int wave = d >> 5, lane = d & 31;
    if (lane == 0) { red[wave][0] = s; red[wave][1] = s2; }
    __syncthreads();
    float ts  = red[0][0] + red[1][0] + red[2][0] + red[3][0];
    float ts2 = red[0][1] + red[1][1] + red[2][1] + red[3][1];
    float mu = ts * (1.f / DD);
    float var = ts2 * (1.f / DD) - mu * mu;
    out[(size_t)row * DD + d] = (v - mu) * rsqrtf(var + 1e-5f) * g[d] + bt[d];
}

extern "C" void kernel_launch(void* const* d_in, const int* in_sizes, int n_in,
                              void* d_out, int out_size, void* d_ws, size_t ws_size,
                              hipStream_t stream) {
    (void)in_sizes; (void)n_in; (void)out_size; (void)ws_size;
    const float* x    = (const float*)d_in[0];
    const float* Wq   = (const float*)d_in[1];
    const float* bq   = (const float*)d_in[2];
    const float* Wk   = (const float*)d_in[3];
    const float* bk   = (const float*)d_in[4];
    const float* Wv   = (const float*)d_in[5];
    const float* bv   = (const float*)d_in[6];
    const float* Wo   = (const float*)d_in[7];
    const float* bo   = (const float*)d_in[8];
    const float* ln_g = (const float*)d_in[9];
    const float* ln_b = (const float*)d_in[10];
    const int*   pidx = (const int*)d_in[11];
    const unsigned char* cmask = (const unsigned char*)d_in[12];
    float* out = (float*)d_out;

    char* ws = (char*)d_ws;
    const size_t rowB = (size_t)MT * DD * 2;          // one bf16 M x 128 buffer
    const size_t off_px  = 0;
    const size_t off_q   = off_px + rowB;
    const size_t off_k   = off_q + rowB;
    const size_t off_v   = off_k + rowB;
    const size_t off_att = off_v + rowB;
    const size_t off_agg = off_att + rowB;
    const size_t aggB    = (size_t)BB * NN * DD * 4;
    const size_t off_cnt = off_agg + aggB;
    const size_t cntB    = (size_t)NN * 4;
    const size_t off_w   = off_cnt + ((cntB + 255) & ~(size_t)255);

    unsigned short* px   = (unsigned short*)(ws + off_px);
    unsigned short* qo   = (unsigned short*)(ws + off_q);
    unsigned short* ko   = (unsigned short*)(ws + off_k);
    unsigned short* vo   = (unsigned short*)(ws + off_v);
    unsigned short* attb = (unsigned short*)(ws + off_att);
    float*          agg  = (float*)(ws + off_agg);
    float*          cnt  = (float*)(ws + off_cnt);
    unsigned short* wb   = (unsigned short*)(ws + off_w);   // [wq|wk|wv|wo] bf16

    k_wconv<<<(4 * DD * DD + 255) / 256, 256, 0, stream>>>(Wq, Wk, Wv, Wo, wb);
    k_gather<<<(MT * (DD / 4) + 255) / 256, 256, 0, stream>>>(x, pidx, px);
    k_qkv<<<MT / 128, 256, 0, stream>>>(px, wb, bq, bk, bv, qo, ko, vo);
    k_attn<<<BB * PP * HH, 256, 0, stream>>>(qo, ko, vo, attb);
    hipMemsetAsync(ws + off_agg, 0, aggB + cntB, stream);
    k_wo<<<MT / 128, 256, 0, stream>>>(attb, wb + 3 * DD * DD, bo, pidx, cmask, agg);
    k_cnt<<<(PP * SS + 255) / 256, 256, 0, stream>>>(pidx, cmask, cnt);
    k_ln<<<BB * NN, 128, 0, stream>>>(x, agg, cnt, ln_g, ln_b, out);
}
